// SNNModel_41403484733458
// MI455X (gfx1250) — compile-verified, matrix operand order checked by reading, not run on hardware
//
#include <hip/hip_runtime.h>

// ---------------------------------------------------------------------------
// SNN 3-layer LIF forward, bf16 WMMA (gfx1250 / MI455X).
// One fused GEMM+LIF kernel per (timestep, layer); mem state in d_ws.
// ---------------------------------------------------------------------------

typedef __bf16 bf16;
typedef __attribute__((ext_vector_type(16))) __bf16 v16bf;
typedef __attribute__((ext_vector_type(8)))  __bf16 v8bf;
typedef __attribute__((ext_vector_type(8)))  float  v8f;

__device__ __forceinline__ v16bf ldf(const bf16* p) {
    // Fragment layout (16-bit A 16x32):
    // lanes 0-15 hold K 0..7 (elems 0..7) and K 16..23 (elems 8..15);
    // lanes 16-31 hold K 8..15 and K 24..31 (caller adds +8 elem offset).
    v8bf lo = *(const v8bf*)(p);
    v8bf hi = *(const v8bf*)(p + 16);
    v16bf r;
#pragma unroll
    for (int i = 0; i < 8; ++i) { r[i] = lo[i]; r[i + 8] = hi[i]; }
    return r;
}

__device__ __forceinline__ v8f wmma_bf16(v16bf a, v16bf b, v8f c) {
    // D = A(16x32) * B(32x16) + C, f32 accumulate
    return __builtin_amdgcn_wmma_f32_16x16x32_bf16(
        /*neg_a=*/false, a, /*neg_b=*/false, b,
        /*c_mod=*/(short)0, c, /*reuse_a=*/false, /*reuse_b=*/false);
}

// Fused GEMM + LIF update.
//   A   : [256, K] bf16, row stride lda
//   W   : [N, K]   bf16 row-major (B matrix = W^T, one W row per B column/lane)
//   mem : [256, N] f32 persistent membrane
//   spk : [256, N] bf16 spike output (next layer input), used when !LAST
//   outf: f32 spike output with row stride ldo, used when LAST
template <int K, int N, bool LAST>
__global__ __launch_bounds__(256) void snn_layer(
    const bf16* __restrict__ A, int lda,
    const bf16* __restrict__ W,
    const float* __restrict__ bias,
    float* __restrict__ mem,
    bf16* __restrict__ spk,
    float* __restrict__ outf, int ldo)
{
    const int lane = threadIdx.x & 31;
    const int wave = threadIdx.x >> 5;          // 0..7
    const int rowBase = blockIdx.x * 64 + (wave >> 2) * 32;   // 2 waves in M
    const int colBase = blockIdx.y * 128 + (wave & 3) * 32;   // 4 waves in N

    const int r    = lane & 15;                 // row (A) / W-row (B) within tile
    const int koff = (lane >> 4) << 3;          // +8 K elems for upper half-wave

    const bf16* a0p = A + (size_t)(rowBase + r) * (size_t)lda + koff;
    const bf16* a1p = a0p + (size_t)16 * (size_t)lda;
    const bf16* b0p = W + (size_t)(colBase + r) * (size_t)K + koff;
    const bf16* b1p = b0p + (size_t)16 * (size_t)K;

    v8f zero = {0.f, 0.f, 0.f, 0.f, 0.f, 0.f, 0.f, 0.f};
    v8f acc00 = zero, acc01 = zero, acc10 = zero, acc11 = zero;

#pragma unroll 4
    for (int k = 0; k < K; k += 32) {
        v16bf a0 = ldf(a0p + k);
        v16bf a1 = ldf(a1p + k);
        v16bf b0 = ldf(b0p + k);
        v16bf b1 = ldf(b1p + k);
        acc00 = wmma_bf16(a0, b0, acc00);
        acc01 = wmma_bf16(a0, b1, acc01);
        acc10 = wmma_bf16(a1, b0, acc10);
        acc11 = wmma_bf16(a1, b1, acc11);
    }

    // C/D layout: element v of lane l -> m = v + 8*(l>=16), n = l & 15
    const int mo = (lane >> 4) << 3;
    const int nc = lane & 15;
    v8f accs[2][2] = {{acc00, acc01}, {acc10, acc11}};

#pragma unroll
    for (int mi = 0; mi < 2; ++mi) {
#pragma unroll
        for (int ni = 0; ni < 2; ++ni) {
            const int col  = colBase + ni * 16 + nc;
            const float bv = bias[col];
            const int rowB = rowBase + mi * 16 + mo;
#pragma unroll
            for (int v = 0; v < 8; ++v) {
                const int row    = rowB + v;
                const size_t idx = (size_t)row * (size_t)N + (size_t)col;
                float m = 0.9f * mem[idx] + accs[mi][ni][v] + bv;  // leak + cur
                float s = (m > 1.0f) ? 1.0f : 0.0f;                // fire
                m -= s;                                            // subtract reset
                mem[idx] = m;
                if (LAST) outf[(size_t)row * (size_t)ldo + (size_t)col] = s;
                else      spk[idx] = (bf16)s;
            }
        }
    }
}

__global__ void cvt_bf16(const float* __restrict__ in, bf16* __restrict__ out, size_t n) {
    size_t i      = (size_t)blockIdx.x * blockDim.x + threadIdx.x;
    size_t stride = (size_t)gridDim.x * blockDim.x;
    for (; i < n; i += stride) out[i] = (bf16)in[i];
}

extern "C" void kernel_launch(void* const* d_in, const int* in_sizes, int n_in,
                              void* d_out, int out_size, void* d_ws, size_t ws_size,
                              hipStream_t stream) {
    constexpr int B = 256, T = 100, D0 = 1024, D1 = 2048, D2 = 1024;

    const float* x  = (const float*)d_in[0];
    const float* W0 = (const float*)d_in[1];
    const float* b0 = (const float*)d_in[2];
    const float* W1 = (const float*)d_in[3];
    const float* b1 = (const float*)d_in[4];
    const float* W2 = (const float*)d_in[5];
    const float* b2 = (const float*)d_in[6];
    float* out = (float*)d_out;

    // Workspace carve-out (256B aligned); ~76 MB total.
    char* w = (char*)d_ws;
    auto alloc = [&](size_t bytes) -> char* {
        char* p = w;
        w += (bytes + 255) & ~(size_t)255;
        return p;
    };
    bf16*  xb   = (bf16*)alloc((size_t)B * T * D0 * 2);
    bf16*  W0b  = (bf16*)alloc((size_t)D1 * D0 * 2);
    bf16*  W1b  = (bf16*)alloc((size_t)D1 * D1 * 2);
    bf16*  W2b  = (bf16*)alloc((size_t)D2 * D1 * 2);
    float* mem0 = (float*)alloc((size_t)B * D1 * 4);
    float* mem1 = (float*)alloc((size_t)B * D1 * 4);
    float* mem2 = (float*)alloc((size_t)B * D2 * 4);
    bf16*  spk0 = (bf16*)alloc((size_t)B * D1 * 2);
    bf16*  spk1 = (bf16*)alloc((size_t)B * D1 * 2);

    // Deterministic state: membranes start at zero every call.
    hipMemsetAsync(mem0, 0, (size_t)B * D1 * 4, stream);
    hipMemsetAsync(mem1, 0, (size_t)B * D1 * 4, stream);
    hipMemsetAsync(mem2, 0, (size_t)B * D2 * 4, stream);

    // One-time per-call f32 -> bf16 conversion (weights L2-resident afterwards).
    cvt_bf16<<<2048, 256, 0, stream>>>(x,  xb,  (size_t)B * T * D0);
    cvt_bf16<<<1024, 256, 0, stream>>>(W0, W0b, (size_t)D1 * D0);
    cvt_bf16<<<1024, 256, 0, stream>>>(W1, W1b, (size_t)D1 * D1);
    cvt_bf16<<<1024, 256, 0, stream>>>(W2, W2b, (size_t)D2 * D1);

    dim3 blk(256);
    dim3 g01(B / 64, D1 / 128);  // (4,16) for N=2048
    dim3 g2 (B / 64, D2 / 128);  // (4,8)  for N=1024

    for (int t = 0; t < T; ++t) {
        // Layer 0: A = x[:, t, :] (bf16 view, row stride T*D0)
        snn_layer<D0, D1, false><<<g01, blk, 0, stream>>>(
            xb + (size_t)t * D0, T * D0, W0b, b0, mem0, spk0, nullptr, 0);
        // Layer 1: A = spk0
        snn_layer<D1, D1, false><<<g01, blk, 0, stream>>>(
            spk0, D1, W1b, b1, mem1, spk1, nullptr, 0);
        // Layer 2: A = spk1, spikes straight to d_out[:, t, :]
        snn_layer<D1, D2, true><<<g2, blk, 0, stream>>>(
            spk1, D1, W2b, b2, mem2, nullptr, out + (size_t)t * D2, T * D2);
    }
}
